// GAT_52329881534972
// MI455X (gfx1250) — matrix operand order repaired and checked by reading, compiled, not run on hardware
//
#include <hip/hip_runtime.h>
#include <hip/hip_bf16.h>

typedef __attribute__((ext_vector_type(16))) _Float16 v16h;
typedef __attribute__((ext_vector_type(8)))  float    v8f;
typedef long long i64;

#define NEG_SLOPE 0.2f
#define DIMH 128

// -------------------- small utility kernels --------------------

__global__ void fill_f32(float* __restrict__ p, float v, long long n) {
    long long i = (long long)blockIdx.x * blockDim.x + threadIdx.x;
    if (i < n) p[i] = v;
}

// float atomic max via int/uint trick (init memory to -inf bit pattern)
__device__ __forceinline__ void atomicMaxF(float* addr, float v) {
    if (v >= 0.0f) atomicMax((int*)addr, __float_as_int(v));
    else           atomicMin((unsigned int*)addr, __float_as_uint(v));
}

// -------------------- weight packing into WMMA B-fragment layout --------------------
// Fragment (ks, nt) covers K rows [ks*32, ks*32+32) x cols [nt*16, nt*16+16).
// Per CDNA5 16-bit B 32x16 layout: lane L holds column nt*16 + (L&15);
// elements e0..e7 = B[ks*32 + (L>=16 ? 8 : 0) + e][col], e8..e15 = B[ks*32 + 16 + (L>=16 ? 8 : 0) + e-8][col].
// Stored contiguously: Bp[((nt*ksteps + ks)*32 + lane)*16 + e]  (32B per lane -> one v16h load)
__global__ void pack_weight(const float* __restrict__ W, _Float16* __restrict__ out,
                            int K, int Ncols, int NcolsReal) {
    int t = blockIdx.x * blockDim.x + threadIdx.x;
    int ksteps = K >> 5;
    int nfrag = ksteps * (Ncols >> 4);
    if (t >= nfrag * 32) return;
    int lane = t & 31;
    int frag = t >> 5;
    int ks = frag % ksteps;
    int nt = frag / ksteps;
    int col = nt * 16 + (lane & 15);
    int kbase = ks * 32 + ((lane >> 4) << 3);
    _Float16* dst = out + (size_t)t * 16;
    for (int e = 0; e < 16; ++e) {
        float v = (col < NcolsReal) ? W[(size_t)(kbase + e + (e & 8)) * NcolsReal + col] : 0.0f;
        dst[e] = (_Float16)v;
    }
}

// -------------------- WMMA GEMM: C[MxN] = A[MxK] * Bpacked, optional bias+ReLU -------
// Block = 128 threads = 4 waves. Wave w owns rows [blockIdx.x*64 + w*16, +16) and
// NT 16-col tiles starting at blockIdx.y*(NT*16).  Requires M % 16 == 0 and
// Ncols % (NT*16) == 0 (true for all call sites).  All WMMA-relevant control flow is
// wave-uniform so EXEC stays all-ones.
template <int NT, bool BIAS, bool RELU>
__global__ void gemm_wmma(const float* __restrict__ A, const _Float16* __restrict__ Bp,
                          float* __restrict__ C, int M, int K, int Ncols,
                          const float* __restrict__ bias) {
    const int lane = threadIdx.x & 31;
    const int wave = threadIdx.x >> 5;
    const int rowbase = blockIdx.x * 64 + wave * 16;
    if (rowbase >= M) return;                  // wave-uniform
    const int colblock = blockIdx.y * (NT * 16);
    const int ksteps = K >> 5;
    const int hi = lane >> 4;
    const int row = rowbase + (lane & 15);
    const float* Arow = A + (size_t)row * K;
    // base of this block's first B fragment: frag index = nt*ksteps + ks
    const int nt0 = colblock >> 4;

    v8f acc[NT] = {};

    for (int ks = 0; ks < ksteps; ++ks) {
        // A fragment: 16-bit A 16x32 layout. Lane<16: e0..7=K0..7, e8..15=K16..23;
        // lane>=16: e0..7=K8..15, e8..15=K24..31.  Two 32B regions, f32->f16 convert.
        int k0 = ks * 32 + hi * 8;
        float4 f0 = ((const float4*)(Arow + k0))[0];
        float4 f1 = ((const float4*)(Arow + k0))[1];
        float4 f2 = ((const float4*)(Arow + k0 + 16))[0];
        float4 f3 = ((const float4*)(Arow + k0 + 16))[1];
        v16h a;
        a[0]=(_Float16)f0.x;  a[1]=(_Float16)f0.y;  a[2]=(_Float16)f0.z;  a[3]=(_Float16)f0.w;
        a[4]=(_Float16)f1.x;  a[5]=(_Float16)f1.y;  a[6]=(_Float16)f1.z;  a[7]=(_Float16)f1.w;
        a[8]=(_Float16)f2.x;  a[9]=(_Float16)f2.y;  a[10]=(_Float16)f2.z; a[11]=(_Float16)f2.w;
        a[12]=(_Float16)f3.x; a[13]=(_Float16)f3.y; a[14]=(_Float16)f3.z; a[15]=(_Float16)f3.w;

#pragma unroll
        for (int t = 0; t < NT; ++t) {
            const _Float16* bp =
                Bp + (((size_t)(nt0 + t) * ksteps + ks) * 32 + lane) * 16;
            v16h b = *(const v16h*)bp;
            acc[t] = __builtin_amdgcn_wmma_f32_16x16x32_f16(
                false, a, false, b, (short)0, acc[t], false, false);
        }
    }

    // C/D layout: element r -> row rowbase + hi*8 + r, col ct + (lane&15)
    const int r0 = rowbase + hi * 8;
#pragma unroll
    for (int t = 0; t < NT; ++t) {
        int col = colblock + t * 16 + (lane & 15);
        float bv = BIAS ? bias[col] : 0.0f;
#pragma unroll
        for (int r = 0; r < 8; ++r) {
            float v = acc[t][r] + bv;
            if (RELU) v = fmaxf(v, 0.0f);
            C[(size_t)(r0 + r) * Ncols + col] = v;
        }
    }
}

// -------------------- per-node attention coefficients (wave per node) ----------------
__global__ void alpha_kernel(const float* __restrict__ H, const float* __restrict__ avs,
                             const float* __restrict__ avd, float* __restrict__ aS,
                             float* __restrict__ aD, int Nn) {
    int lane = threadIdx.x & 31;
    int node = (int)(((long long)blockIdx.x * blockDim.x + threadIdx.x) >> 5);
    if (node >= Nn) return;
    float4 h  = ((const float4*)(H + (size_t)node * DIMH))[lane];
    float4 s4 = ((const float4*)avs)[lane];
    float4 d4 = ((const float4*)avd)[lane];
    float ps = h.x * s4.x + h.y * s4.y + h.z * s4.z + h.w * s4.w;
    float pd = h.x * d4.x + h.y * d4.y + h.z * d4.z + h.w * d4.w;
    for (int off = 16; off > 0; off >>= 1) {
        ps += __shfl_xor(ps, off, 32);
        pd += __shfl_xor(pd, off, 32);
    }
    if (lane == 0) { aS[node] = ps; aD[node] = pd; }
}

// -------------------- edge softmax passes --------------------
__device__ __forceinline__ void edge_sd(const i64* ei, int E, int e, int& s, int& d) {
    if (e < E) { s = (int)ei[e]; d = (int)ei[(size_t)E + e]; }
    else       { s = d = e - E; }               // self loops appended
}

__global__ void edge_logit_max(const i64* __restrict__ ei, int E, int Nn,
                               const float* __restrict__ aS, const float* __restrict__ aD,
                               float* __restrict__ logit, float* __restrict__ m) {
    int e = blockIdx.x * blockDim.x + threadIdx.x;
    if (e >= E + Nn) return;
    int s, d; edge_sd(ei, E, e, s, d);
    float x = aS[s] + aD[d];
    float l = x > 0.0f ? x : NEG_SLOPE * x;
    logit[e] = l;
    atomicMaxF(m + d, l);
}

__global__ void edge_exp_sum(const i64* __restrict__ ei, int E, int Nn,
                             const float* __restrict__ m, float* __restrict__ logit_ex,
                             float* __restrict__ denom) {
    int e = blockIdx.x * blockDim.x + threadIdx.x;
    if (e >= E + Nn) return;
    int s, d; edge_sd(ei, E, e, s, d);
    float ex = __expf(logit_ex[e] - m[d]);
    logit_ex[e] = ex;
    atomicAdd(denom + d, ex);
}

// wave per edge: 32 lanes x float4 = 128 features; atomic scatter-add resolves in L2
__global__ void edge_aggregate(const i64* __restrict__ ei, int E, int Nn,
                               const float* __restrict__ ex, const float* __restrict__ denom,
                               const float* __restrict__ H, float* __restrict__ agg) {
    int lane = threadIdx.x & 31;
    int e = (int)(((long long)blockIdx.x * blockDim.x + threadIdx.x) >> 5);
    if (e >= E + Nn) return;
    int s, d; edge_sd(ei, E, e, s, d);
    float alpha = ex[e] / (denom[d] + 1e-16f);
    float4 h = ((const float4*)(H + (size_t)s * DIMH))[lane];
    float* dst = agg + (size_t)d * DIMH + lane * 4;
    atomicAdd(dst + 0, alpha * h.x);
    atomicAdd(dst + 1, alpha * h.y);
    atomicAdd(dst + 2, alpha * h.z);
    atomicAdd(dst + 3, alpha * h.w);
}

__global__ void bias_relu(const float* __restrict__ agg, const float* __restrict__ bias,
                          float* __restrict__ out, long long n) {
    long long i = (long long)blockIdx.x * blockDim.x + threadIdx.x;
    if (i >= n) return;
    out[i] = fmaxf(agg[i] + bias[(int)(i & (DIMH - 1))], 0.0f);
}

// -------------------- global mean pool --------------------
__global__ void pool_sum(const float* __restrict__ Hf, const i64* __restrict__ batch,
                         float* __restrict__ sums, long long n) {
    long long i = (long long)blockIdx.x * blockDim.x + threadIdx.x;
    if (i >= n) return;
    int node = (int)(i >> 7), c = (int)(i & 127);
    atomicAdd(sums + (size_t)((int)batch[node]) * DIMH + c, Hf[i]);
}

__global__ void pool_cnt(const i64* __restrict__ batch, float* __restrict__ cnt, int Nn) {
    int i = blockIdx.x * blockDim.x + threadIdx.x;
    if (i < Nn) atomicAdd(cnt + (int)batch[i], 1.0f);
}

__global__ void pool_div(const float* __restrict__ sums, const float* __restrict__ cnt,
                         float* __restrict__ pooled, int n) {
    int i = blockIdx.x * blockDim.x + threadIdx.x;
    if (i >= n) return;
    pooled[i] = sums[i] / fmaxf(cnt[i >> 7], 1.0f);
}

__global__ void final_out(const float* __restrict__ Cpad, const float* __restrict__ bf3,
                          float* __restrict__ out, int n) {
    int i = blockIdx.x * blockDim.x + threadIdx.x;
    if (i >= n) return;
    int g = i / 10, c = i - g * 10;
    out[i] = Cpad[g * 16 + c] + bf3[c];
}

// -------------------- host orchestration --------------------

extern "C" void kernel_launch(void* const* d_in, const int* in_sizes, int n_in,
                              void* d_out, int out_size, void* d_ws, size_t ws_size,
                              hipStream_t stream) {
    const float* x    = (const float*)d_in[0];
    const i64*   ei   = (const i64*)d_in[1];
    const i64*   batch= (const i64*)d_in[2];
    const float* W1   = (const float*)d_in[3];
    const float* as1  = (const float*)d_in[4];
    const float* ad1  = (const float*)d_in[5];
    const float* b1   = (const float*)d_in[6];
    const float* W2   = (const float*)d_in[7];
    const float* as2  = (const float*)d_in[8];
    const float* ad2  = (const float*)d_in[9];
    const float* b2   = (const float*)d_in[10];
    const float* W3   = (const float*)d_in[11];
    const float* as3  = (const float*)d_in[12];
    const float* ad3  = (const float*)d_in[13];
    const float* b3   = (const float*)d_in[14];
    const float* Wf1  = (const float*)d_in[15];
    const float* bf1  = (const float*)d_in[16];
    const float* Wf2  = (const float*)d_in[17];
    const float* bf2  = (const float*)d_in[18];
    const float* Wf3  = (const float*)d_in[19];
    const float* bf3  = (const float*)d_in[20];

    const int Nn   = in_sizes[0] / DIMH;   // 50000
    const int E    = in_sizes[1] / 2;      // 640000
    const int Etot = E + Nn;
    const long long NF = (long long)Nn * DIMH;

    // workspace carve-out (256B aligned)
    char* base = (char*)d_ws;
    size_t cur = 0;
    auto alloc = [&](size_t bytes) -> void* {
        void* p = base + cur;
        cur = (cur + bytes + 255) & ~(size_t)255;
        return p;
    };
    float*    Xw   = (float*)alloc(NF * 4);
    float*    Hh   = (float*)alloc(NF * 4);
    float*    AGG  = (float*)alloc(NF * 4);
    float*    aS   = (float*)alloc((size_t)Nn * 4);
    float*    aD   = (float*)alloc((size_t)Nn * 4);
    float*    Mx   = (float*)alloc((size_t)Nn * 4);
    float*    Dn   = (float*)alloc((size_t)Nn * 4);
    float*    Elog = (float*)alloc((size_t)Etot * 4);
    _Float16* Wp1  = (_Float16*)alloc(4 * 8  * 32 * 16 * 2);
    _Float16* Wp2  = (_Float16*)alloc(4 * 8  * 32 * 16 * 2);
    _Float16* Wp3  = (_Float16*)alloc(4 * 8  * 32 * 16 * 2);
    _Float16* Wf1p = (_Float16*)alloc(4 * 32 * 32 * 16 * 2);
    _Float16* Wf2p = (_Float16*)alloc(16 * 32 * 32 * 16 * 2);
    _Float16* Wf3p = (_Float16*)alloc(16 * 1  * 32 * 16 * 2);
    float*    sums = (float*)alloc(128 * DIMH * 4);
    float*    cnt  = (float*)alloc(128 * 4);
    float*    pooled = (float*)alloc(128 * DIMH * 4);
    float*    z1   = (float*)alloc(128 * 512 * 4);
    float*    z2   = (float*)alloc(128 * 512 * 4);
    float*    Cpad = (float*)alloc(128 * 16 * 4);

    auto packW = [&](const float* W, _Float16* out, int K, int Nc, int NcReal) {
        int threads = (K / 32) * (Nc / 16) * 32;
        pack_weight<<<(threads + 127) / 128, 128, 0, stream>>>(W, out, K, Nc, NcReal);
    };
    // wide path: wave covers 16 rows x 128 cols (8 tiles); Ncols % 128 == 0
    auto gemmWide = [&](const float* A, const _Float16* Bp, float* C, int M, int K, int Nc,
                        const float* bias, bool relu) {
        dim3 grid((M + 63) / 64, Nc / 128);
        if (relu) gemm_wmma<8, true, true><<<grid, 128, 0, stream>>>(A, Bp, C, M, K, Nc, bias);
        else      gemm_wmma<8, false, false><<<grid, 128, 0, stream>>>(A, Bp, C, M, K, Nc, nullptr);
    };
    auto gemmNarrow = [&](const float* A, const _Float16* Bp, float* C, int M, int K, int Nc) {
        dim3 grid((M + 63) / 64, Nc / 16);
        gemm_wmma<1, false, false><<<grid, 128, 0, stream>>>(A, Bp, C, M, K, Nc, nullptr);
    };

    packW(W1, Wp1, 128, 128, 128);
    packW(W2, Wp2, 128, 128, 128);
    packW(W3, Wp3, 128, 128, 128);
    packW(Wf1, Wf1p, 128, 512, 512);
    packW(Wf2, Wf2p, 512, 512, 512);
    packW(Wf3, Wf3p, 512, 16, 10);

    auto gatLayer = [&](const float* Xin, const _Float16* Wpk,
                        const float* avs, const float* avd, const float* bb) {
        gemmWide(Xin, Wpk, Hh, Nn, DIMH, DIMH, nullptr, false);
        alpha_kernel<<<(int)(((long long)Nn * 32 + 127) / 128), 128, 0, stream>>>(
            Hh, avs, avd, aS, aD, Nn);
        fill_f32<<<(Nn + 255) / 256, 256, 0, stream>>>(Mx, -__builtin_inff(), Nn);
        fill_f32<<<(Nn + 255) / 256, 256, 0, stream>>>(Dn, 0.0f, Nn);
        fill_f32<<<(int)((NF + 255) / 256), 256, 0, stream>>>(AGG, 0.0f, NF);
        edge_logit_max<<<(Etot + 255) / 256, 256, 0, stream>>>(ei, E, Nn, aS, aD, Elog, Mx);
        edge_exp_sum<<<(Etot + 255) / 256, 256, 0, stream>>>(ei, E, Nn, Mx, Elog, Dn);
        edge_aggregate<<<(int)(((long long)Etot * 32 + 255) / 256), 256, 0, stream>>>(
            ei, E, Nn, Elog, Dn, Hh, AGG);
        bias_relu<<<(int)((NF + 255) / 256), 256, 0, stream>>>(AGG, bb, Xw, NF);
    };

    gatLayer(x,  Wp1, as1, ad1, b1);
    gatLayer(Xw, Wp2, as2, ad2, b2);
    gatLayer(Xw, Wp3, as3, ad3, b3);

    // global mean pool
    fill_f32<<<(128 * DIMH + 255) / 256, 256, 0, stream>>>(sums, 0.0f, 128 * DIMH);
    fill_f32<<<1, 128, 0, stream>>>(cnt, 0.0f, 128);
    pool_sum<<<(int)((NF + 255) / 256), 256, 0, stream>>>(Xw, batch, sums, NF);
    pool_cnt<<<(Nn + 255) / 256, 256, 0, stream>>>(batch, cnt, Nn);
    pool_div<<<(128 * DIMH + 255) / 256, 256, 0, stream>>>(sums, cnt, pooled, 128 * DIMH);

    // FFN head
    gemmWide(pooled, Wf1p, z1, 128, 128, 512, bf1, true);
    gemmWide(z1,     Wf2p, z2, 128, 512, 512, bf2, true);
    gemmNarrow(z2,   Wf3p, Cpad, 128, 512, 16);
    final_out<<<(128 * 10 + 255) / 256, 256, 0, stream>>>(Cpad, bf3, (float*)d_out, 128 * 10);
}
